// ScaledDotProductAttention_70858370449594
// MI455X (gfx1250) — matrix operand order repaired and checked by reading, not compiled, not run on hardware
//
#include <hip/hip_runtime.h>

// Problem constants (from reference): B=2, H=16, S=2048, D=64, causal mask.
#define BATCH  2
#define HEADS  16
#define SLEN   2048
#define DHEAD  64
#define QTILE  64     // query rows per workgroup
#define KTILE  64     // key columns per iteration
#define NW     4      // waves per workgroup (wave32)
#define NTHR   (NW * 32)

typedef _Float16 h16 __attribute__((ext_vector_type(16)));
typedef float    f8  __attribute__((ext_vector_type(8)));
typedef unsigned int u32x4 __attribute__((ext_vector_type(4)));
typedef int      i32x8 __attribute__((ext_vector_type(8)));
typedef int      i32x4 __attribute__((ext_vector_type(4)));

__device__ __forceinline__ f8 wmma_f16(h16 a, h16 b, f8 c) {
  // (neg_a, A, neg_b, B, c_mod, C, reuse_a, reuse_b)
  return __builtin_amdgcn_wmma_f32_16x16x32_f16(false, a, false, b, (short)0, c,
                                                false, false);
}

__device__ __forceinline__ float red_max16(float x) {
  x = fmaxf(x, __shfl_xor(x, 1));
  x = fmaxf(x, __shfl_xor(x, 2));
  x = fmaxf(x, __shfl_xor(x, 4));
  x = fmaxf(x, __shfl_xor(x, 8));
  return x;
}
__device__ __forceinline__ float red_sum16(float x) {
  x += __shfl_xor(x, 1);
  x += __shfl_xor(x, 2);
  x += __shfl_xor(x, 4);
  x += __shfl_xor(x, 8);
  return x;
}

// TDM: DMA one fully-in-bounds 64x64 f32 tile (row stride 64 elems) into LDS.
// D# layout per CDNA5 ISA ch.8: group0 {count, lds_addr, global_addr, type=2},
// group1 {data_size=4B, tensor dims/strides, tile dims}, groups 2/3 zero (2-D).
__device__ __forceinline__ void tdm_load_tile_f32(unsigned lds_off,
                                                  const float* gsrc) {
  unsigned long long ga = (unsigned long long)(uintptr_t)gsrc;
  u32x4 g0;
  g0[0] = 1u;                                         // count=1 (valid user D#)
  g0[1] = lds_off;                                    // lds_addr (bytes)
  g0[2] = (unsigned)(ga & 0xFFFFFFFFu);               // global_addr[31:0]
  g0[3] = (unsigned)((ga >> 32) & 0x1FFFFFFu)         // global_addr[56:32]
        | (2u << 30);                                 // type=2 ("image")
  i32x8 g1;
  g1[0] = (2 << 16);          // workgroup_mask=0, data_size=2 -> 4 bytes
  g1[1] = (64 << 16);         // tensor_dim0[15:0]=64  (atomic_barrier_addr=0)
  g1[2] = (64 << 16);         // tensor_dim0[31:16]=0, tensor_dim1[15:0]=64
  g1[3] = (64 << 16);         // tensor_dim1[31:16]=0, tile_dim0=64
  g1[4] = 64;                 // tile_dim1=64, tile_dim2=0 (unused -> 2D)
  g1[5] = 64;                 // tensor_dim0_stride[31:0]=64
  g1[6] = 0;                  // stride hi / tensor_dim1_stride lo
  g1[7] = 0;
  i32x4 z4 = {0, 0, 0, 0};            // groups 2/3: dims 2..4 unused
  i32x8 z8 = {0, 0, 0, 0, 0, 0, 0, 0};
  // amdgpu-toolchain (clang-23) 6-arg form: (g0, g1, g2, g3, g4, cpol)
  __builtin_amdgcn_tensor_load_to_lds(g0, g1, z4, z4, z8, 0);
}

__global__ __launch_bounds__(NTHR, 1)
void fa_fwd_kernel(const float* __restrict__ Q, const float* __restrict__ K,
                   const float* __restrict__ V, float* __restrict__ Out,
                   float* __restrict__ Attn) {
  __shared__ float    Kraw[2][KTILE][DHEAD];  // 32KB double-buffered TDM landing
  __shared__ float    Vraw[2][KTILE][DHEAD];  // 32KB double-buffered TDM landing
  __shared__ _Float16 Ks[KTILE][DHEAD];       // 8KB  K tile f16 [key][d]
  __shared__ _Float16 VTs[DHEAD][KTILE];      // 8KB  V tile f16 transposed [d][key]
  __shared__ _Float16 Ps[NW][16][KTILE];      // 8KB  per-wave P staging (C->A)

  const int qb   = blockIdx.x;                // 0..S/QTILE-1
  const int bh   = blockIdx.y;                // 0..B*H-1
  const int tid  = threadIdx.x;
  const int w    = tid >> 5;
  const int lane = tid & 31;
  const int half = lane >> 4;                 // 0: lanes 0-15, 1: lanes 16-31
  const int l16  = lane & 15;

  const size_t base = (size_t)bh * SLEN * DHEAD;
  const float* q = Q + base;
  const float* k = K + base;
  const float* v = V + base;
  float* o    = Out  + base;
  float* attn = Attn + (size_t)bh * SLEN * SLEN;

  const int qrow0 = qb * QTILE + w * 16;      // wave's first global q row

  const unsigned ldsK0 = (unsigned)(uintptr_t)&Kraw[0][0][0];
  const unsigned ldsK1 = (unsigned)(uintptr_t)&Kraw[1][0][0];
  const unsigned ldsV0 = (unsigned)(uintptr_t)&Vraw[0][0][0];
  const unsigned ldsV1 = (unsigned)(uintptr_t)&Vraw[1][0][0];

  // ---- Load Q stripe into two A operands (16x32 halves of D) ----
  // Fold scale = D^-1/2 = 0.125 (exact power of two) into the f16 Q operand,
  // so WMMA outputs are pre-scaled scores.
  h16 a0, a1;
  {
    const float* qr = q + (size_t)(qrow0 + l16) * DHEAD;
    const int c0 = half * 16;
#pragma unroll
    for (int i = 0; i < 8; ++i) {
      a0[i]     = (_Float16)(qr[c0 + i] * 0.125f);
      a0[i + 8] = (_Float16)(qr[c0 + 8 + i] * 0.125f);
      a1[i]     = (_Float16)(qr[32 + c0 + i] * 0.125f);
      a1[i + 8] = (_Float16)(qr[32 + c0 + 8 + i] * 0.125f);
    }
  }

  // B operand from K tile: B[K=d][N=key], lane col = l16, K halves contiguous.
  auto loadB_K = [&](int ktile16, int dbase) -> h16 {
    const _Float16* p = &Ks[ktile16 + l16][dbase + half * 16];
    h16 b;
#pragma unroll
    for (int e = 0; e < 16; ++e) b[e] = p[e];
    return b;
  };
  // B operand from V tile (transposed): B[K=key][N=d].
  auto loadB_V = [&](int kchunk, int dtile) -> h16 {
    const _Float16* p = &VTs[dtile * 16 + l16][kchunk * 32 + half * 16];
    h16 b;
#pragma unroll
    for (int e = 0; e < 16; ++e) b[e] = p[e];
    return b;
  };

  // f32 LDS raw tile -> f16 operand tiles (K row-major, V transposed).
  auto convert_K = [&](int buf) {
#pragma unroll 1
    for (int i = tid; i < KTILE * DHEAD / 4; i += NTHR) {
      float4 t = ((const float4*)&Kraw[buf][0][0])[i];
      int kk = (i * 4) / DHEAD, dd = (i * 4) % DHEAD;
      Ks[kk][dd + 0] = (_Float16)t.x;
      Ks[kk][dd + 1] = (_Float16)t.y;
      Ks[kk][dd + 2] = (_Float16)t.z;
      Ks[kk][dd + 3] = (_Float16)t.w;
    }
  };
  auto convert_V = [&](int buf) {
#pragma unroll 1
    for (int i = tid; i < KTILE * DHEAD / 4; i += NTHR) {
      float4 t = ((const float4*)&Vraw[buf][0][0])[i];
      int kk = (i * 4) / DHEAD, dd = (i * 4) % DHEAD;
      VTs[dd + 0][kk] = (_Float16)t.x;
      VTs[dd + 1][kk] = (_Float16)t.y;
      VTs[dd + 2][kk] = (_Float16)t.z;
      VTs[dd + 3][kk] = (_Float16)t.w;
    }
  };

  // Scores for one 64-key block: four 16x16 C tiles (pre-scaled, causal-masked).
  auto scores = [&](int kk0, f8 (&c)[4]) {
#pragma unroll
    for (int t = 0; t < 4; ++t) {
      f8 z = {};
      c[t] = wmma_f16(a0, loadB_K(t * 16, 0), z);
      c[t] = wmma_f16(a1, loadB_K(t * 16, 32), c[t]);
    }
#pragma unroll
    for (int t = 0; t < 4; ++t) {
      const int col = kk0 + t * 16 + l16;
#pragma unroll
      for (int r = 0; r < 8; ++r) {
        const int qg = qrow0 + r + half * 8;
        c[t][r] = (col <= qg) ? c[t][r] : -INFINITY;
      }
    }
  };

  float m[8], l[8];
#pragma unroll
  for (int r = 0; r < 8; ++r) { m[r] = -INFINITY; l[r] = 0.0f; }

  const int nblk = qb + 1;                    // causal 64-key blocks for this WG

  // ================= Pass 1: softmax statistics (m, l) =================
  // Double-buffered TDM pipeline: issue j+1 before waiting for j.
  // TDM ops complete in-order per wave, so tensorcnt<=N releases the oldest.
  if (w == 0) tdm_load_tile_f32(ldsK0, k);    // preload block 0
  for (int j = 0; j < nblk; ++j) {
    const int cur = j & 1;
    if (w == 0) {
      if (j + 1 < nblk) {
        tdm_load_tile_f32(cur ? ldsK0 : ldsK1,
                          k + (size_t)((j + 1) * KTILE) * DHEAD);
        __builtin_amdgcn_s_wait_tensorcnt(1);   // oldest (block j) done
      } else {
        __builtin_amdgcn_s_wait_tensorcnt(0);   // drain
      }
    }
    __syncthreads();                          // raw[cur] visible to all waves
    convert_K(cur);
    __syncthreads();                          // f16 tile ready
    f8 c[4];
    scores(j * KTILE, c);
#pragma unroll
    for (int r = 0; r < 8; ++r) {
      float tmax = fmaxf(fmaxf(c[0][r], c[1][r]), fmaxf(c[2][r], c[3][r]));
      tmax = red_max16(tmax);
      float mn = fmaxf(m[r], tmax);
      float alpha = __expf(m[r] - mn);
      float e = __expf(c[0][r] - mn) + __expf(c[1][r] - mn) +
                __expf(c[2][r] - mn) + __expf(c[3][r] - mn);
      l[r] = l[r] * alpha + red_sum16(e);
      m[r] = mn;
    }
    __syncthreads();                          // all waves past Ks before reuse
  }
  float invl[8];
#pragma unroll
  for (int r = 0; r < 8; ++r) invl[r] = 1.0f / l[r];

  // ============ Pass 2: P (normalized) -> attn store + O = P*V ============
  f8 acc[4] = {};
  if (w == 0) {                               // preload block 0 (K and V)
    tdm_load_tile_f32(ldsK0, k);
    tdm_load_tile_f32(ldsV0, v);
  }
  for (int j = 0; j < nblk; ++j) {
    const int cur = j & 1;
    if (w == 0) {
      if (j + 1 < nblk) {
        const size_t off = (size_t)((j + 1) * KTILE) * DHEAD;
        tdm_load_tile_f32(cur ? ldsK0 : ldsK1, k + off);
        tdm_load_tile_f32(cur ? ldsV0 : ldsV1, v + off);
        __builtin_amdgcn_s_wait_tensorcnt(2);   // oldest pair (block j) done
      } else {
        __builtin_amdgcn_s_wait_tensorcnt(0);   // drain
      }
    }
    __syncthreads();
    convert_K(cur);
    convert_V(cur);
    __syncthreads();
    const int kk0 = j * KTILE;
    f8 c[4];
    scores(kk0, c);
#pragma unroll
    for (int t = 0; t < 4; ++t) {
#pragma unroll
      for (int r = 0; r < 8; ++r) {
        const int qg = qrow0 + r + half * 8;
        float p = __expf(c[t][r] - m[r]) * invl[r];   // exp(-inf)=0 for mask
        attn[(size_t)qg * SLEN + kk0 + t * 16 + l16] = p;
        Ps[w][r + half * 8][t * 16 + l16] = (_Float16)p;
      }
    }
    // Reload P in A layout (wave-private LDS; DS ops in-order per wave).
    h16 ap0, ap1;
    {
      const _Float16* pr = &Ps[w][l16][0];
      const int c0i = half * 16;
#pragma unroll
      for (int i = 0; i < 8; ++i) {
        ap0[i]     = pr[c0i + i];
        ap0[i + 8] = pr[c0i + 8 + i];
        ap1[i]     = pr[32 + c0i + i];
        ap1[i + 8] = pr[32 + c0i + 8 + i];
      }
    }
#pragma unroll
    for (int t = 0; t < 4; ++t) {
      acc[t] = wmma_f16(ap0, loadB_V(0, t), acc[t]);
      acc[t] = wmma_f16(ap1, loadB_V(1, t), acc[t]);
    }
    __syncthreads();                          // all waves past Ks/VTs before reuse
  }

  // ---- Store O (C layout -> [q][d]) ----
#pragma unroll
  for (int r = 0; r < 8; ++r) {
    const int qg = qrow0 + r + half * 8;
    float* orow = o + (size_t)qg * DHEAD;
#pragma unroll
    for (int t = 0; t < 4; ++t) orow[t * 16 + l16] = acc[t][r];
  }

  // ---- Zero-fill fully-masked attention columns [ (qb+1)*64, S ) ----
  const int zstart = (qb + 1) * QTILE;
  const int zcols = SLEN - zstart;
  if (zcols > 0) {
    const int vper = zcols / 4;
    const float4 zero = {0.f, 0.f, 0.f, 0.f};
#pragma unroll 1
    for (int i = tid; i < QTILE * vper; i += NTHR) {
      int rr = i / vper;
      int cc = (i % vper) * 4;
      float4* dst = (float4*)(attn + (size_t)(qb * QTILE + rr) * SLEN + zstart + cc);
      *dst = zero;
    }
  }
}

extern "C" void kernel_launch(void* const* d_in, const int* in_sizes, int n_in,
                              void* d_out, int out_size, void* d_ws, size_t ws_size,
                              hipStream_t stream) {
  const float* Q = (const float*)d_in[0];
  const float* K = (const float*)d_in[1];
  const float* V = (const float*)d_in[2];
  // d_in[3] is the causal mask; causality is applied analytically in-kernel.
  float* Out  = (float*)d_out;
  float* Attn = Out + (size_t)BATCH * HEADS * SLEN * DHEAD;

  dim3 grid(SLEN / QTILE, BATCH * HEADS);
  dim3 block(NTHR);
  fa_fwd_kernel<<<grid, block, 0, stream>>>(Q, K, V, Out, Attn);
}
